// LlamaTransformer_46136538693947
// MI455X (gfx1250) — matrix operand order repaired and checked by reading, not compiled, not run on hardware
//
#include <hip/hip_runtime.h>
#include <hip/hip_bf16.h>

// ---------------------------------------------------------------------------
// Llama-style transformer forward for gfx1250 (MI455X).
// GEMMs run through v_wmma_f32_16x16x32_bf16. The main GEMM stages its A tile
// into LDS with global_load_async_to_lds_b128 (CDNA5 async path, ASYNCcnt) and
// its B tile transposed as bf16 in LDS so WMMA fragment reads are contiguous
// ds_load_b128s. Fragment layouts follow CDNA5 ISA §7.12.2.
// ---------------------------------------------------------------------------

typedef __attribute__((ext_vector_type(16))) __bf16 bf16x16;
typedef __attribute__((ext_vector_type(8)))  float  f32x8;

#define DIM     1024
#define S_LEN   2048
#define NTOK    4096      // B*S = 2*2048
#define NH      16
#define NKV     4
#define HD      64
#define NE      8
#define HIDDEN  2816
#define VOCABN  32000
#define NEG_INF (-1.0e30f)

// Flip to 0 if the async-to-LDS mnemonic is rejected by the assembler.
#define USE_ASYNC_COPY 1

__device__ __forceinline__ f32x8 wmma_bf16(bf16x16 a, bf16x16 b, f32x8 c) {
  return __builtin_amdgcn_wmma_f32_16x16x32_bf16(
      /*neg_a=*/false, a, /*neg_b=*/false, b,
      /*c_mod=*/(short)0, c, /*reuse_a=*/false, /*reuse_b=*/false);
}

#if USE_ASYNC_COPY
// Per-lane async copy of 16 bytes global -> LDS (tracked by ASYNCcnt).
// vdst = LDS byte address VGPR, vaddr = 64-bit global address VGPR pair.
__device__ __forceinline__ void async_ld_b128(unsigned lds_addr, const float* gaddr) {
  asm volatile("global_load_async_to_lds_b128 %0, %1, off"
               :: "v"(lds_addr), "v"((unsigned long long)(uintptr_t)gaddr)
               : "memory");
}
__device__ __forceinline__ void wait_asynccnt0() {
  asm volatile("s_wait_asynccnt 0x0" ::: "memory");
}
#endif

// ---------------------------------------------------------------------------
// Block-tiled GEMM: C[M,N] (+)= row_scale(row) * (A[M,K] @ B[K,N]).
// 256 threads = 8 waves in a 4x2 grid; block tile 64x128, wave tile 16x64,
// K-step 32 (one WMMA K). M%64==0, N%128==0, K%32==0 at every call site.
// ---------------------------------------------------------------------------
__global__ void __launch_bounds__(256)
gemm_bf16_wmma(const float* __restrict__ A,
               const float* __restrict__ B,
               float* __restrict__ C,
               int M, int N, int K, int accumulate,
               const float* __restrict__ row_scale,
               int rs_stride, int rs_off) {
  __shared__ float  Af[64 * 32];     // A tile, f32 (8 KB)
  __shared__ __bf16 Bs[128 * 32];    // B tile, bf16, TRANSPOSED [n][k] (8 KB)

  const int tid  = threadIdx.x;
  const int lane = tid & 31;
  const int wid  = tid >> 5;

  const int tiles_n = N >> 7;                  // N / 128
  const int bm = blockIdx.x / tiles_n;
  const int bn = blockIdx.x % tiles_n;
  const int row0g = bm << 6;                   // block row base
  const int nb    = bn << 7;                   // block col base

  const int wm = wid & 3;                      // wave m index 0..3
  const int wn = wid >> 2;                     // wave n index 0..1
  const int wrow = wm << 4;                    // wave row offset within tile
  const int wcol = wn << 6;                    // wave col offset within tile

  f32x8 acc[4] = {{}, {}, {}, {}};

  // Per-thread staging assignments (fixed across K-steps):
  //   A: 2048 f32 = 512 float4 chunks; 2 per thread.
  //   B: 4096 f32 = thread covers row (tid>>3), 16 cols at ((tid&7)<<4).
  const int brow = tid >> 3;
  const int bc0  = (tid & 7) << 4;

  for (int k0 = 0; k0 < K; k0 += 32) {
    __syncthreads();                           // previous compute done

    // ---- stage A tile: global -> LDS (f32), async path ----
#pragma unroll
    for (int c = 0; c < 2; ++c) {
      const int chunk = tid * 2 + c;
      const int arow  = chunk >> 3;            // 8 float4 per 32-wide row
      const int acol  = (chunk & 7) << 2;
      const float* gp = A + (size_t)(row0g + arow) * K + k0 + acol;
#if USE_ASYNC_COPY
      async_ld_b128((unsigned)(uintptr_t)&Af[arow * 32 + acol], gp);
#else
      const float4 v = *(const float4*)gp;
      float* lp = &Af[arow * 32 + acol];
      lp[0] = v.x; lp[1] = v.y; lp[2] = v.z; lp[3] = v.w;
#endif
    }

    // ---- stage B tile: global f32 -> LDS bf16, transposed [n][k] ----
    {
      const float4* gp = (const float4*)(B + (size_t)(k0 + brow) * N + nb + bc0);
#pragma unroll
      for (int j = 0; j < 4; ++j) {
        const float4 v = gp[j];
        const int cb = bc0 + (j << 2);
        Bs[(cb + 0) * 32 + brow] = (__bf16)v.x;
        Bs[(cb + 1) * 32 + brow] = (__bf16)v.y;
        Bs[(cb + 2) * 32 + brow] = (__bf16)v.z;
        Bs[(cb + 3) * 32 + brow] = (__bf16)v.w;
      }
    }

#if USE_ASYNC_COPY
    wait_asynccnt0();
#endif
    __syncthreads();                           // tiles visible to all waves

    // ---- compute: 1 A frag (reused 4x), 4 B frags, 4 WMMA ----
    bf16x16 a;
    {
      const int row = wrow + (lane & 15);
      const int kb  = (lane & 16) ? 8 : 0;
      const float* p = &Af[row * 32 + kb];
#pragma unroll
      for (int i = 0; i < 8; ++i) { a[i] = (__bf16)p[i]; a[8 + i] = (__bf16)p[16 + i]; }
    }
#pragma unroll
    for (int t = 0; t < 4; ++t) {
      const int col = wcol + (t << 4) + (lane & 15);
      const int kk  = (lane & 16) ? 8 : 0;
      const __bf16* p = &Bs[col * 32 + kk];
      bf16x16 bfr;
#pragma unroll
      for (int i = 0; i < 8; ++i) { bfr[i] = p[i]; bfr[8 + i] = p[16 + i]; }
      acc[t] = wmma_bf16(a, bfr, acc[t]);
    }
  }

  // ---- store: C layout = rows over VGPRs, cols over lanes ----
  const int rbase = row0g + wrow + ((lane & 16) ? 8 : 0);
  const int cbase = nb + wcol + (lane & 15);
#pragma unroll
  for (int r = 0; r < 8; ++r) {
    const int row = rbase + r;
    const float sc = row_scale ? row_scale[(size_t)row * rs_stride + rs_off] : 1.0f;
    float* cp = C + (size_t)row * N + cbase;
    if (accumulate) {
      cp[0]  += sc * acc[0][r];
      cp[16] += sc * acc[1][r];
      cp[32] += sc * acc[2][r];
      cp[48] += sc * acc[3][r];
    } else {
      cp[0]  = sc * acc[0][r];
      cp[16] = sc * acc[1][r];
      cp[32] = sc * acc[2][r];
      cp[48] = sc * acc[3][r];
    }
  }
}

// ---------------------------------------------------------------------------
// Direct-from-global fragment loaders (used by the attention kernel).
// ---------------------------------------------------------------------------
__device__ __forceinline__ bf16x16
frag_a_rowmajor(const float* base, int ld, int row0, int k0, int lane) {
  const float* p = base + (size_t)(row0 + (lane & 15)) * ld + k0 + ((lane & 16) ? 8 : 0);
  bf16x16 f;
#pragma unroll
  for (int i = 0; i < 8; ++i) { f[i] = (__bf16)p[i]; f[8 + i] = (__bf16)p[16 + i]; }
  return f;
}

__device__ __forceinline__ bf16x16
frag_b_rowmajor(const float* base, int ld, int k0, int n0, int lane) {
  const float* p = base + (size_t)(k0 + ((lane & 16) ? 8 : 0)) * ld + n0 + (lane & 15);
  bf16x16 f;
#pragma unroll
  for (int i = 0; i < 8; ++i) { f[i] = (__bf16)p[(size_t)i * ld]; f[8 + i] = (__bf16)p[(size_t)(16 + i) * ld]; }
  return f;
}

__device__ __forceinline__ bf16x16
frag_b_transposed(const float* base, int ld, int k0, int n0, int lane) {
  const float* p = base + (size_t)(n0 + (lane & 15)) * ld + k0 + ((lane & 16) ? 8 : 0);
  bf16x16 f;
#pragma unroll
  for (int i = 0; i < 8; ++i) { f[i] = (__bf16)p[i]; f[8 + i] = (__bf16)p[16 + i]; }
  return f;
}

// ---------------------------------------------------------------------------
// Embedding gather: h[t,:] = tok_emb[tokens[t],:]
// ---------------------------------------------------------------------------
__global__ void embed_kernel(const int* __restrict__ tokens,
                             const float* __restrict__ tok_emb,
                             float* __restrict__ h, int total) {
  int idx = blockIdx.x * blockDim.x + threadIdx.x;
  if (idx >= total) return;
  int t = idx >> 10;
  int d = idx & (DIM - 1);
  h[idx] = tok_emb[(size_t)tokens[t] * DIM + d];
}

// ---------------------------------------------------------------------------
// RMSNorm: one 256-thread block per row.
// ---------------------------------------------------------------------------
__global__ void rmsnorm_kernel(const float* __restrict__ x,
                               const float* __restrict__ w,
                               float* __restrict__ y, int D) {
  const int row = blockIdx.x;
  const float* xr = x + (size_t)row * D;
  float ss = 0.f;
  for (int i = threadIdx.x; i < D; i += blockDim.x) { float v = xr[i]; ss += v * v; }
#pragma unroll
  for (int off = 16; off > 0; off >>= 1) ss += __shfl_xor(ss, off, 32);
  __shared__ float sred[8];
  if ((threadIdx.x & 31) == 0) sred[threadIdx.x >> 5] = ss;
  __syncthreads();
  if (threadIdx.x == 0) {
    float t = 0.f;
#pragma unroll
    for (int i = 0; i < 8; ++i) t += sred[i];
    sred[0] = rsqrtf(t / (float)D + 1e-6f);
  }
  __syncthreads();
  const float r = sred[0];
  float* yr = y + (size_t)row * D;
  for (int i = threadIdx.x; i < D; i += blockDim.x) yr[i] = xr[i] * r * w[i];
}

// ---------------------------------------------------------------------------
// RoPE (in place)
// ---------------------------------------------------------------------------
__global__ void rope_kernel(float* __restrict__ x, const int* __restrict__ start_pos,
                            int n_heads, int total) {
  int idx = blockIdx.x * blockDim.x + threadIdx.x;
  if (idx >= total) return;
  int i   = idx & 31;                         // HD/2 = 32
  int hh  = (idx >> 5) % n_heads;
  int tok = idx / (32 * n_heads);
  int s   = tok & (S_LEN - 1);
  float pos = (float)(start_pos[0] + s);
  float inv = __expf(-((float)(2 * i) / (float)HD) * __logf(10000.0f));
  float ang = pos * inv;
  float c = __cosf(ang), sn = __sinf(ang);
  float* p = x + (size_t)tok * (n_heads * HD) + hh * HD + 2 * i;
  float x1 = p[0], x2 = p[1];
  p[0] = x1 * c - x2 * sn;
  p[1] = x1 * sn + x2 * c;
}

// ---------------------------------------------------------------------------
// Flash-style causal attention. One wave per (batch, head, 16-row q block).
// QK^T and PV both via WMMA; P goes through LDS (C-layout -> A-layout).
// ---------------------------------------------------------------------------
__global__ void attn_kernel(const float* __restrict__ q,
                            const float* __restrict__ k,
                            const float* __restrict__ v,
                            float* __restrict__ o) {
  __shared__ float pbuf[16 * 32];
  const int blk  = blockIdx.x;
  const int qb   = blk & 127;                  // S/16
  const int h    = (blk >> 7) & (NH - 1);
  const int b    = blk >> 11;
  const int lane = threadIdx.x & 31;
  const int g    = h >> 2;                     // kv head (N_REP = 4)

  const float* qbase = q + (size_t)b * S_LEN * DIM + h * HD;
  const float* kbase = k + (size_t)b * S_LEN * (NKV * HD) + g * HD;
  const float* vbase = v + (size_t)b * S_LEN * (NKV * HD) + g * HD;
  const int q0 = qb << 4;

  bf16x16 qa0 = frag_a_rowmajor(qbase, DIM, q0, 0,  lane);
  bf16x16 qa1 = frag_a_rowmajor(qbase, DIM, q0, 32, lane);

  f32x8 o0 = {}, o1 = {}, o2 = {}, o3 = {};
  float m_run[8], l_run[8];
#pragma unroll
  for (int r = 0; r < 8; ++r) { m_run[r] = NEG_INF; l_run[r] = 0.f; }

  const int rlocal = (lane & 16) ? 8 : 0;
  const int col    = lane & 15;
  const float scale = 0.125f;                  // 1/sqrt(HD)

  const int kv_end = q0 + 16;
  for (int kv0 = 0; kv0 < kv_end; kv0 += 32) {
    f32x8 s0 = {}, s1 = {};
    {
      bf16x16 kb = frag_b_transposed(kbase, NKV * HD, 0,  kv0,      lane);
      s0 = wmma_bf16(qa0, kb, s0);
      kb = frag_b_transposed(kbase, NKV * HD, 32, kv0,      lane);
      s0 = wmma_bf16(qa1, kb, s0);
      kb = frag_b_transposed(kbase, NKV * HD, 0,  kv0 + 16, lane);
      s1 = wmma_bf16(qa0, kb, s1);
      kb = frag_b_transposed(kbase, NKV * HD, 32, kv0 + 16, lane);
      s1 = wmma_bf16(qa1, kb, s1);
    }

    float mnew[8];
#pragma unroll
    for (int r = 0; r < 8; ++r) {
      const int qi = q0 + rlocal + r;
      float a = s0[r] * scale, bb = s1[r] * scale;
      if (kv0 + col      > qi) a  = NEG_INF;
      if (kv0 + 16 + col > qi) bb = NEG_INF;
      s0[r] = a; s1[r] = bb;
      float v2 = fmaxf(a, bb);
#pragma unroll
      for (int off = 8; off > 0; off >>= 1) v2 = fmaxf(v2, __shfl_xor(v2, off, 32));
      mnew[r] = fmaxf(m_run[r], v2);
    }
#pragma unroll
    for (int r = 0; r < 8; ++r) {
      const float alpha = __expf(m_run[r] - mnew[r]);
      m_run[r] = mnew[r];
      l_run[r] *= alpha;
      o0[r] *= alpha; o1[r] *= alpha; o2[r] *= alpha; o3[r] *= alpha;
      float p0 = __expf(s0[r] - m_run[r]);
      float p1 = __expf(s1[r] - m_run[r]);
      float rs = p0 + p1;
#pragma unroll
      for (int off = 8; off > 0; off >>= 1) rs += __shfl_xor(rs, off, 32);
      l_run[r] += rs;
      pbuf[(rlocal + r) * 32 + col]      = p0;
      pbuf[(rlocal + r) * 32 + 16 + col] = p1;
    }
    __syncthreads();

    bf16x16 pa;
    {
      const float* pp = &pbuf[(lane & 15) * 32 + ((lane & 16) ? 8 : 0)];
#pragma unroll
      for (int i = 0; i < 8; ++i) {
        pa[i]     = (__bf16)pp[i];
        pa[8 + i] = (__bf16)pp[16 + i];
      }
    }
    o0 = wmma_bf16(pa, frag_b_rowmajor(vbase, NKV * HD, kv0, 0,  lane), o0);
    o1 = wmma_bf16(pa, frag_b_rowmajor(vbase, NKV * HD, kv0, 16, lane), o1);
    o2 = wmma_bf16(pa, frag_b_rowmajor(vbase, NKV * HD, kv0, 32, lane), o2);
    o3 = wmma_bf16(pa, frag_b_rowmajor(vbase, NKV * HD, kv0, 48, lane), o3);
    __syncthreads();
  }

#pragma unroll
  for (int r = 0; r < 8; ++r) {
    const int qi = q0 + rlocal + r;
    const float inv_l = 1.0f / l_run[r];
    float* op = o + (size_t)(b * S_LEN + qi) * DIM + h * HD + col;
    op[0]  = o0[r] * inv_l;
    op[16] = o1[r] * inv_l;
    op[32] = o2[r] * inv_l;
    op[48] = o3[r] * inv_l;
  }
}

// ---------------------------------------------------------------------------
// MoE gate: softmax over 8 experts, top-2, normalized weights -> cw[t, 8].
// ---------------------------------------------------------------------------
__global__ void gate_topk_kernel(const float* __restrict__ xn,
                                 const float* __restrict__ gw,
                                 float* __restrict__ cw) {
  const int t = blockIdx.x;
  const int lane = threadIdx.x & 31;
  const float* xr = xn + (size_t)t * DIM;
  float acc[NE];
#pragma unroll
  for (int e = 0; e < NE; ++e) acc[e] = 0.f;
  for (int d = lane; d < DIM; d += 32) {
    const float xv = xr[d];
#pragma unroll
    for (int e = 0; e < NE; ++e) acc[e] += xv * gw[(size_t)d * NE + e];
  }
#pragma unroll
  for (int e = 0; e < NE; ++e) {
#pragma unroll
    for (int off = 16; off > 0; off >>= 1) acc[e] += __shfl_xor(acc[e], off, 32);
  }
  if (lane == 0) {
    float mx = acc[0];
#pragma unroll
    for (int e = 1; e < NE; ++e) mx = fmaxf(mx, acc[e]);
    float pr[NE], se = 0.f;
#pragma unroll
    for (int e = 0; e < NE; ++e) { pr[e] = __expf(acc[e] - mx); se += pr[e]; }
#pragma unroll
    for (int e = 0; e < NE; ++e) pr[e] /= se;
    int i1 = 0;
#pragma unroll
    for (int e = 1; e < NE; ++e) if (pr[e] > pr[i1]) i1 = e;
    int i2 = (i1 == 0) ? 1 : 0;
#pragma unroll
    for (int e = 0; e < NE; ++e) if (e != i1 && pr[e] > pr[i2]) i2 = e;
    const float s2 = pr[i1] + pr[i2];
#pragma unroll
    for (int e = 0; e < NE; ++e)
      cw[(size_t)t * NE + e] = (e == i1) ? pr[i1] / s2 : ((e == i2) ? pr[i2] / s2 : 0.f);
  }
}

// h1 = silu(h1) * h3
__global__ void silu_mul_kernel(float* __restrict__ h1,
                                const float* __restrict__ h3, int total) {
  int idx = blockIdx.x * blockDim.x + threadIdx.x;
  if (idx >= total) return;
  const float x = h1[idx];
  h1[idx] = (x / (1.0f + __expf(-x))) * h3[idx];
}

// ---------------------------------------------------------------------------
// Host-side orchestration
// ---------------------------------------------------------------------------
static inline void launch_gemm(hipStream_t s, const float* A, const float* B, float* C,
                               int M, int N, int K, int acc,
                               const float* rs, int rss, int rso) {
  const int blocks = (M >> 6) * (N >> 7);      // 64x128 block tiles
  gemm_bf16_wmma<<<blocks, 256, 0, s>>>(A, B, C, M, N, K, acc, rs, rss, rso);
}

extern "C" void kernel_launch(void* const* d_in, const int* in_sizes, int n_in,
                              void* d_out, int out_size, void* d_ws, size_t ws_size,
                              hipStream_t stream) {
  (void)in_sizes; (void)n_in; (void)out_size; (void)ws_size;

  const int*   tokens      = (const int*)  d_in[0];
  const int*   start_pos   = (const int*)  d_in[1];
  const float* tok_emb     = (const float*)d_in[2];
  const float* attn_norm_w = (const float*)d_in[3];
  const float* wq          = (const float*)d_in[4];
  const float* wk          = (const float*)d_in[5];
  const float* wv          = (const float*)d_in[6];
  const float* wo          = (const float*)d_in[7];
  const float* ffn_norm_w  = (const float*)d_in[8];
  const float* gate_w      = (const float*)d_in[9];
  const float* w1          = (const float*)d_in[10];
  const float* w2          = (const float*)d_in[11];
  const float* w3          = (const float*)d_in[12];
  const float* norm_w      = (const float*)d_in[13];
  const float* out_w       = (const float*)d_in[14];
  float* logits            = (float*)d_out;

  float* ws = (float*)d_ws;
  float* h    = ws;                 ws += (size_t)NTOK * DIM;
  float* xn   = ws;                 ws += (size_t)NTOK * DIM;
  float* qb   = ws;                 ws += (size_t)NTOK * DIM;
  float* kb   = ws;                 ws += (size_t)NTOK * NKV * HD;
  float* vb   = ws;                 ws += (size_t)NTOK * NKV * HD;
  float* h1   = ws;                 ws += (size_t)NTOK * HIDDEN;
  float* h3   = ws;                 ws += (size_t)NTOK * HIDDEN;
  float* cw   = ws;                 ws += (size_t)NTOK * NE;

  {
    const int total = NTOK * DIM;
    embed_kernel<<<(total + 255) / 256, 256, 0, stream>>>(tokens, tok_emb, h, total);
  }

  for (int l = 0; l < 2; ++l) {
    const float* wq_l = wq + (size_t)l * DIM * DIM;
    const float* wk_l = wk + (size_t)l * DIM * (NKV * HD);
    const float* wv_l = wv + (size_t)l * DIM * (NKV * HD);
    const float* wo_l = wo + (size_t)l * DIM * DIM;
    const float* gw_l = gate_w + (size_t)l * DIM * NE;

    rmsnorm_kernel<<<NTOK, 256, 0, stream>>>(h, attn_norm_w + (size_t)l * DIM, xn, DIM);
    launch_gemm(stream, xn, wq_l, qb, NTOK, DIM,      DIM, 0, nullptr, 0, 0);
    launch_gemm(stream, xn, wk_l, kb, NTOK, NKV * HD, DIM, 0, nullptr, 0, 0);
    launch_gemm(stream, xn, wv_l, vb, NTOK, NKV * HD, DIM, 0, nullptr, 0, 0);
    {
      const int tq = NTOK * NH * (HD / 2);
      rope_kernel<<<(tq + 255) / 256, 256, 0, stream>>>(qb, start_pos, NH, tq);
      const int tk = NTOK * NKV * (HD / 2);
      rope_kernel<<<(tk + 255) / 256, 256, 0, stream>>>(kb, start_pos, NKV, tk);
    }
    attn_kernel<<<2 * NH * (S_LEN / 16), 32, 0, stream>>>(qb, kb, vb, xn);
    launch_gemm(stream, xn, wo_l, h, NTOK, DIM, DIM, 1, nullptr, 0, 0);

    rmsnorm_kernel<<<NTOK, 256, 0, stream>>>(h, ffn_norm_w + (size_t)l * DIM, xn, DIM);
    gate_topk_kernel<<<NTOK, 32, 0, stream>>>(xn, gw_l, cw);
    for (int e = 0; e < NE; ++e) {
      const float* w1_le = w1 + ((size_t)l * NE + e) * DIM * HIDDEN;
      const float* w3_le = w3 + ((size_t)l * NE + e) * DIM * HIDDEN;
      const float* w2_le = w2 + ((size_t)l * NE + e) * HIDDEN * DIM;
      launch_gemm(stream, xn, w1_le, h1, NTOK, HIDDEN, DIM, 0, nullptr, 0, 0);
      launch_gemm(stream, xn, w3_le, h3, NTOK, HIDDEN, DIM, 0, nullptr, 0, 0);
      {
        const int total = NTOK * HIDDEN;
        silu_mul_kernel<<<(total + 255) / 256, 256, 0, stream>>>(h1, h3, total);
      }
      launch_gemm(stream, h1, w2_le, h, NTOK, DIM, HIDDEN, 1, cw, NE, e);
    }
  }

  rmsnorm_kernel<<<NTOK, 256, 0, stream>>>(h, norm_w, xn, DIM);
  launch_gemm(stream, xn, out_w, logits, NTOK, VOCABN, DIM, 0, nullptr, 0, 0);
}
